// GraphEscapeScorer_43430709297913
// MI455X (gfx1250) — matrix operand order repaired and checked by reading, not compiled, not run on hardware
//
#include <hip/hip_runtime.h>
#include <hip/hip_bf16.h>

// Problem constants (from reference): N,E,B,H,P,ET,L
constexpr int cN = 8192;
constexpr int cE = 65536;
constexpr int cB = 2;
constexpr int cH = 192;
constexpr int cL = 4;

typedef __attribute__((ext_vector_type(16))) __bf16 bf16x16;
typedef __attribute__((ext_vector_type(8)))  __bf16 bf16x8;
typedef __attribute__((ext_vector_type(8)))  float  floatx8;

// ------------------------------------------------------------------ utilities
__global__ void fill_kernel(float* __restrict__ p, float v, int n) {
  int i = blockIdx.x * 256 + threadIdx.x;
  if (i < n) p[i] = v;
}

__device__ __forceinline__ float siluf(float x) { return x / (1.f + __expf(-x)); }

__device__ __forceinline__ void atomicMaxF(float* addr, float val) {
  unsigned int* ua = (unsigned int*)addr;
  unsigned int old = *ua;
  while (true) {
    float f = __uint_as_float(old);
    if (f >= val) break;
    unsigned int assumed = old;
    old = atomicCAS(ua, assumed, __float_as_uint(val));
    if (old == assumed) break;
  }
}

// Pre-swizzle fp32 weights [K][Nout] into per-lane WMMA B-fragment order (bf16):
// dst[((kt*(Nout/16) + nt)*32 + lane)*16 + j] = W[kt*32 + khalf*16 + j][nt*16 + ncol]
// with lane = khalf*16 + ncol.  In the GEMM each lane reads its 32-byte fragment
// as one aligned bf16x16 (-> 2x global_load_b128, fully coalesced, L2 resident).
__global__ void pack_w_kernel(const float* __restrict__ W, __bf16* __restrict__ Wpk,
                              int K, int Nout) {
  int idx = blockIdx.x * 256 + threadIdx.x;
  if (idx >= K * Nout) return;
  int k = idx / Nout, n = idx % Nout;
  int kt = k >> 5, kl = k & 31, khalf = kl >> 4, j = kl & 15;
  int nt = n >> 4, ncol = n & 15;
  int lane = khalf * 16 + ncol;
  size_t dst = (((size_t)kt * (Nout >> 4) + nt) * 32 + lane) * 16 + j;
  Wpk[dst] = (__bf16)W[idx];
}

// --------------------------------------------------------------- WMMA GEMM
// C[M][Nout] = A[M][K](bf16) @ W[K][Nout](bf16, pre-swizzled) + bias, opt SiLU.
// 128 threads = 4 waves; block computes a 64x64 output tile (4 M-tiles per wave,
// reusing each B fragment across 4 wmma).  K-step 64 => 8 wmma per iteration.
template <bool SILU, typename OutT>
__global__ void __launch_bounds__(128)
gemm_bias_kernel(const __bf16* __restrict__ A,
                 const __bf16* __restrict__ Wpk,
                 const float*  __restrict__ bias,
                 OutT* __restrict__ C,
                 int M, int K, int Nout) {
  // Two 64x32 A tiles; rows padded to 40 elems: 16B-aligned rows; within any
  // 16-row fragment group the start banks (20*r mod 64) are all distinct ->
  // conflict-free b128 reads (group offset 320*mt == 0 mod 64).
  __shared__ __align__(16) __bf16 As[2][64][40];

  const int m0    = blockIdx.x * 64;
  const int wave  = threadIdx.x >> 5;
  const int lane  = threadIdx.x & 31;
  const int ntile = blockIdx.y * 4 + wave;
  const int nT    = Nout >> 4;
  const int lrow  = lane & 15;
  const int khalf = lane >> 4;

  floatx8 acc[4] = {{}, {}, {}, {}};
  const int nkt = K >> 5;  // 32-wide K tiles
  for (int kt = 0; kt < nkt; kt += 2) {
    // stage two 64x32 A tiles: 512 16B-chunks, 4 per thread
    bf16x8 av[4];
#pragma unroll
    for (int j = 0; j < 4; ++j) {
      int cidx = threadIdx.x + 128 * j;
      int lt = cidx >> 8, rem = cidx & 255;
      int row = rem >> 2, col = (rem & 3) * 8;
      av[j] = *(const bf16x8*)(A + (size_t)(m0 + row) * K + (kt + lt) * 32 + col);
    }
    __syncthreads();  // previous iteration's readers done
#pragma unroll
    for (int j = 0; j < 4; ++j) {
      int cidx = threadIdx.x + 128 * j;
      int lt = cidx >> 8, rem = cidx & 255;
      int row = rem >> 2, col = (rem & 3) * 8;
      *(bf16x8*)(&As[lt][row][col]) = av[j];
    }
    __syncthreads();
#pragma unroll
    for (int t = 0; t < 2; ++t) {
      // B fragment: contiguous 32B per lane from pre-swizzled weights
      bf16x16 bfv = *(const bf16x16*)(
          Wpk + (((size_t)(kt + t) * nT + ntile) * 32 + lane) * 16);
#pragma unroll
      for (int mt = 0; mt < 4; ++mt) {
        // A fragment: lanes 0-15 K{0..15}, lanes 16-31 K{16..31}
        bf16x8 alo = *(const bf16x8*)(&As[t][mt * 16 + lrow][khalf * 16]);
        bf16x8 ahi = *(const bf16x8*)(&As[t][mt * 16 + lrow][khalf * 16 + 8]);
        bf16x16 af;
#pragma unroll
        for (int e = 0; e < 8; ++e) { af[e] = alo[e]; af[8 + e] = ahi[e]; }
        acc[mt] = __builtin_amdgcn_wmma_f32_16x16x32_bf16(
            /*neg_a=*/false, af, /*neg_b=*/false, bfv,
            /*c_mod=*/(short)0, acc[mt], /*reuse_a=*/false, /*reuse_b=*/false);
      }
    }
  }

  // D layout: VGPR r -> (M = mtile*16 + r + (lane>=16 ? 8 : 0), N = ntile*16 + lane%16)
  const int nc   = ntile * 16 + lrow;
  const float bv = bias[nc];
#pragma unroll
  for (int mt = 0; mt < 4; ++mt) {
    const int rbase = m0 + mt * 16 + (khalf << 3);
#pragma unroll
    for (int r = 0; r < 8; ++r) {
      float x = acc[mt][r] + bv;
      if (SILU) x = siluf(x);
      C[(size_t)(rbase + r) * Nout + nc] = (OutT)x;
    }
  }
}

// ----------------------------------------------------------- graph pipeline
__global__ void __launch_bounds__(256)
stats_kernel(const float* __restrict__ srcm, const float* __restrict__ candm,
             float* __restrict__ stats) {
  int b = blockIdx.x, t = threadIdx.x;
  float s0 = 0, s1 = 0, s2 = 0, s3 = 0, s4 = 0;
  for (int n = t; n < cN; n += 256) {
    float sm = srcm[b * cN + n], cm = candm[b * cN + n];
    s0 += sm; s1 += cm; s2 += fabsf(cm - sm);
    s3 += fmaxf(sm, 0.f); s4 += fmaxf(cm, 0.f);
  }
  __shared__ float r[5][256];
  r[0][t] = s0; r[1][t] = s1; r[2][t] = s2; r[3][t] = s3; r[4][t] = s4;
  __syncthreads();
  for (int s = 128; s > 0; s >>= 1) {
    if (t < s)
      for (int i = 0; i < 5; ++i) r[i][t] += r[i][t + s];
    __syncthreads();
  }
  if (t < 5) stats[b * 8 + t] = r[t][0];
}

// stage 1 of encoder: t = silu(enc1([points, node_features])) -> bf16
__global__ void enc1_kernel(const float* __restrict__ pts,
                            const float* __restrict__ srcm,
                            const float* __restrict__ candm,
                            const float* __restrict__ stats,
                            const float* __restrict__ w,   // (11, H)
                            const float* __restrict__ bias,
                            __bf16* __restrict__ out) {
  size_t idx = (size_t)blockIdx.x * 256 + threadIdx.x;
  if (idx >= (size_t)cB * cN * cH) return;
  int h = idx % cH;
  size_t bn = idx / cH;
  int n = bn % cN, b = bn / cN;
  float sm = srcm[b * cN + n], cm = candm[b * cN + n];
  float f[11];
  f[0] = pts[n * 3 + 0]; f[1] = pts[n * 3 + 1]; f[2] = pts[n * 3 + 2];
  f[3] = sm; f[4] = cm; f[5] = 1.f - sm; f[6] = 1.f - cm;
  f[7] = cm - sm; f[8] = fabsf(cm - sm);
  f[9] = stats[b * 8 + 0] / (float)cN;
  f[10] = stats[b * 8 + 1] / (float)cN;
  float s = bias[h];
#pragma unroll
  for (int i = 0; i < 11; ++i) s += f[i] * w[i * cH + h];
  out[idx] = (__bf16)siluf(s);
}

// gather per-edge activations: A_tgt = hidden[tgt]; A_src = hidden[src]+edge_emb
__global__ void edge_prep_kernel(const float* __restrict__ hidden,
                                 const int* __restrict__ eidx,
                                 const int* __restrict__ etype,
                                 const float* __restrict__ edge_emb,
                                 __bf16* __restrict__ a_tgt,
                                 __bf16* __restrict__ a_src) {
  size_t idx = (size_t)blockIdx.x * 256 + threadIdx.x;
  if (idx >= (size_t)cB * cE * cH) return;
  int h = idx % cH;
  size_t be = idx / cH;
  int e = be % cE, b = be / cE;
  int s = eidx[e], t = eidx[cE + e];
  a_tgt[idx] = (__bf16)hidden[((size_t)b * cN + t) * cH + h];
  a_src[idx] = (__bf16)(hidden[((size_t)b * cN + s) * cH + h] +
                        edge_emb[(size_t)etype[e] * cH + h]);
}

// per-edge bias logit: ebl[e] = eb(edge_emb[type[e]])
__global__ void ebl_kernel(const float* __restrict__ edge_emb,
                           const int* __restrict__ etype,
                           const float* __restrict__ ebw,
                           const float* __restrict__ ebb,
                           float* __restrict__ ebl) {
  int e = blockIdx.x * 256 + threadIdx.x;
  if (e >= cE) return;
  const float* em = edge_emb + (size_t)etype[e] * cH;
  float s = ebb[0];
  for (int h = 0; h < cH; ++h) s += em[h] * ebw[h];
  ebl[e] = s;
}

// logits = dot(q,k)/sqrt(H) + ebl  (one wave per (b,e))
__global__ void __launch_bounds__(256)
logits_kernel(const __bf16* __restrict__ q, const __bf16* __restrict__ k,
              const float* __restrict__ ebl, float* __restrict__ logits) {
  int wid = (int)(((size_t)blockIdx.x * 256 + threadIdx.x) >> 5);
  int lane = threadIdx.x & 31;
  if (wid >= cB * cE) return;
  const __bf16* qp = q + (size_t)wid * cH;
  const __bf16* kp = k + (size_t)wid * cH;
  float s = 0.f;
  for (int h = lane; h < cH; h += 32) s += (float)qp[h] * (float)kp[h];
  for (int off = 16; off > 0; off >>= 1) s += __shfl_down(s, off, 32);
  if (lane == 0)
    logits[wid] = s * 0.07216878364870323f + ebl[wid % cE];  // 1/sqrt(192)
}

__global__ void segmax_kernel(const float* __restrict__ logits,
                              const int* __restrict__ eidx,
                              float* __restrict__ m) {
  int idx = blockIdx.x * 256 + threadIdx.x;
  if (idx >= cB * cE) return;
  int e = idx % cE, b = idx / cE;
  atomicMaxF(&m[b * cN + eidx[cE + e]], logits[idx]);
}

__global__ void expsum_kernel(float* __restrict__ logits,
                              const int* __restrict__ eidx,
                              const float* __restrict__ m,
                              float* __restrict__ ssum) {
  int idx = blockIdx.x * 256 + threadIdx.x;
  if (idx >= cB * cE) return;
  int e = idx % cE, b = idx / cE;
  int t = eidx[cE + e];
  float ex = __expf(logits[idx] - m[b * cN + t]);
  logits[idx] = ex;
  atomicAdd(&ssum[b * cN + t], ex);
}

__global__ void agg_kernel(const __bf16* __restrict__ v,
                           const float* __restrict__ ex,
                           const float* __restrict__ ssum,
                           const int* __restrict__ eidx,
                           float* __restrict__ agg) {
  size_t idx = (size_t)blockIdx.x * 256 + threadIdx.x;
  if (idx >= (size_t)cB * cE * cH) return;
  int h = idx % cH;
  size_t be = idx / cH;
  int e = be % cE, b = be / cE;
  int t = eidx[cE + e];
  float w = ex[be] / ssum[b * cN + t];
  atomicAdd(&agg[((size_t)b * cN + t) * cH + h], (float)v[idx] * w);
}

__global__ void ucat_kernel(const float* __restrict__ hidden,
                            const float* __restrict__ agg,
                            __bf16* __restrict__ ucat) {
  size_t idx = (size_t)blockIdx.x * 256 + threadIdx.x;
  if (idx >= (size_t)cB * cN * 2 * cH) return;
  int h2 = idx % (2 * cH);
  size_t bn = idx / (2 * cH);
  float x = (h2 < cH) ? hidden[bn * cH + h2] : agg[bn * cH + (h2 - cH)];
  ucat[idx] = (__bf16)x;
}

// hidden = LayerNorm(hidden + upd) * g + b  (block per node)
__global__ void __launch_bounds__(256)
resln_kernel(float* __restrict__ hidden, const float* __restrict__ upd,
             const float* __restrict__ g, const float* __restrict__ beta) {
  size_t node = blockIdx.x;
  int t = threadIdx.x;
  __shared__ float red[256];
  float x = 0.f;
  if (t < cH) x = hidden[node * cH + t] + upd[node * cH + t];
  red[t] = x;
  __syncthreads();
  for (int s = 128; s > 0; s >>= 1) { if (t < s) red[t] += red[t + s]; __syncthreads(); }
  float mu = red[0] / (float)cH;
  __syncthreads();
  float d = (t < cH) ? (x - mu) : 0.f;
  red[t] = d * d;
  __syncthreads();
  for (int s = 128; s > 0; s >>= 1) { if (t < s) red[t] += red[t + s]; __syncthreads(); }
  float inv = rsqrtf(red[0] / (float)cH + 1e-5f);
  if (t < cH) hidden[node * cH + t] = g[t] * (x - mu) * inv + beta[t];
}

// pooled: [0]=ss, [1]=cs, [2]=gs each (B,H); block per (b,h)
__global__ void __launch_bounds__(256)
pool_kernel(const float* __restrict__ hidden, const float* __restrict__ srcm,
            const float* __restrict__ candm, const float* __restrict__ stats,
            float* __restrict__ pooled) {
  int b = blockIdx.x / cH, h = blockIdx.x % cH;
  float s1 = 0, s2 = 0, s3 = 0;
  for (int n = threadIdx.x; n < cN; n += 256) {
    float hv = hidden[((size_t)b * cN + n) * cH + h];
    s1 += fmaxf(srcm[b * cN + n], 0.f) * hv;
    s2 += fmaxf(candm[b * cN + n], 0.f) * hv;
    s3 += hv;
  }
  __shared__ float r1[256], r2[256], r3[256];
  int t = threadIdx.x;
  r1[t] = s1; r2[t] = s2; r3[t] = s3;
  __syncthreads();
  for (int s = 128; s > 0; s >>= 1) {
    if (t < s) { r1[t] += r1[t + s]; r2[t] += r2[t + s]; r3[t] += r3[t + s]; }
    __syncthreads();
  }
  if (t == 0) {
    float ds = fmaxf(stats[b * 8 + 3], 1e-6f);
    float dc = fmaxf(stats[b * 8 + 4], 1e-6f);
    pooled[(0 * cB + b) * cH + h] = r1[0] / ds;
    pooled[(1 * cB + b) * cH + h] = r2[0] / dc;
    pooled[(2 * cB + b) * cH + h] = r3[0] / (float)cN;
  }
}

// tiny head: x(772) -> silu(head1) -> head2 ; block per batch
__global__ void __launch_bounds__(256)
head_kernel(const float* __restrict__ pooled, const float* __restrict__ stats,
            const float* __restrict__ energy,
            const float* __restrict__ w1, const float* __restrict__ b1,
            const float* __restrict__ w2, const float* __restrict__ b2,
            float* __restrict__ out) {
  int b = blockIdx.x, t = threadIdx.x;
  __shared__ float sh[192];
  __shared__ float red[256];
  if (t < cH) {
    float s = b1[t];
    for (int i = 0; i < 4 * cH + 4; ++i) {
      float x;
      if (i < cH)            x = pooled[(2 * cB + b) * cH + i];           // gs
      else if (i < 2 * cH)   x = pooled[(0 * cB + b) * cH + (i - cH)];    // ss
      else if (i < 3 * cH)   x = pooled[(1 * cB + b) * cH + (i - 2*cH)];  // cs
      else if (i < 4 * cH) { int j = i - 3 * cH;
        x = pooled[(1 * cB + b) * cH + j] - pooled[(0 * cB + b) * cH + j];// ds
      } else {
        int j = i - 4 * cH;
        x = (j == 0) ? stats[b * 8 + 2] / (float)cN
          : (j == 1) ? stats[b * 8 + 1]
          : (j == 2) ? stats[b * 8 + 0]
          : energy[b];
      }
      s += x * w1[(size_t)i * cH + t];
    }
    sh[t] = siluf(s);
  }
  __syncthreads();
  red[t] = (t < cH) ? sh[t] * w2[t] : 0.f;
  __syncthreads();
  for (int s = 128; s > 0; s >>= 1) { if (t < s) red[t] += red[t + s]; __syncthreads(); }
  if (t == 0) out[b] = red[0] + b2[0];
}

// ------------------------------------------------------------------- launch
struct Lin { const float* w; const float* b; };

extern "C" void kernel_launch(void* const* d_in, const int* in_sizes, int n_in,
                              void* d_out, int out_size, void* d_ws, size_t ws_size,
                              hipStream_t stream) {
  const float* points = (const float*)d_in[0];
  const float* srcm   = (const float*)d_in[1];
  const float* candm  = (const float*)d_in[2];
  const int*   eidx   = (const int*)d_in[3];
  const int*   etype  = (const int*)d_in[4];
  const float* energy = (const float*)d_in[5];

  auto gf = [&](int i) { return (const float*)d_in[i]; };

  Lin enc1{}, enc2{}, head1{}, head2{};
  const float* edge_emb[cL]; Lin q[cL], k[cL], v[cL], eb[cL], u1[cL], u2[cL];
  const float* ln_g[cL]; const float* ln_b[cL];

  // Disambiguate pytree flattening: sorted keys (jax default) puts enc1.b
  // (size 192) at index 6; insertion order puts enc1.w (size 2112) there.
  bool sorted_mode = (in_sizes[6] == 192);
  if (sorted_mode) {
    enc1.b = gf(6);  enc1.w = gf(7);
    enc2.b = gf(8);  enc2.w = gf(9);
    head1.b = gf(10); head1.w = gf(11);
    head2.b = gf(12); head2.w = gf(13);
    int base = 14;
    for (int l = 0; l < cL; ++l) {
      eb[l].b = gf(base + 0);  eb[l].w = gf(base + 1);
      edge_emb[l] = gf(base + 2);
      k[l].b = gf(base + 3);   k[l].w = gf(base + 4);
      ln_b[l] = gf(base + 5);  ln_g[l] = gf(base + 6);
      q[l].b = gf(base + 7);   q[l].w = gf(base + 8);
      u1[l].b = gf(base + 9);  u1[l].w = gf(base + 10);
      u2[l].b = gf(base + 11); u2[l].w = gf(base + 12);
      v[l].b = gf(base + 13);  v[l].w = gf(base + 14);
      base += 15;
    }
  } else {
    enc1.w = gf(6);  enc1.b = gf(7);
    enc2.w = gf(8);  enc2.b = gf(9);
    int base = 10;
    for (int l = 0; l < cL; ++l) {
      edge_emb[l] = gf(base + 0);
      q[l].w = gf(base + 1);  q[l].b = gf(base + 2);
      k[l].w = gf(base + 3);  k[l].b = gf(base + 4);
      v[l].w = gf(base + 5);  v[l].b = gf(base + 6);
      eb[l].w = gf(base + 7); eb[l].b = gf(base + 8);
      u1[l].w = gf(base + 9); u1[l].b = gf(base + 10);
      u2[l].w = gf(base + 11); u2[l].b = gf(base + 12);
      ln_g[l] = gf(base + 13); ln_b[l] = gf(base + 14);
      base += 15;
    }
    head1.w = gf(base + 0); head1.b = gf(base + 1);
    head2.w = gf(base + 2); head2.b = gf(base + 3);
  }

  // workspace carving
  char* wp = (char*)d_ws;
  auto carve = [&](size_t bytes) -> void* {
    void* r = (void*)wp;
    wp += (bytes + 255) & ~(size_t)255;
    return r;
  };
  const size_t BN = (size_t)cB * cN, BE = (size_t)cB * cE;
  float*  stats  = (float*) carve(cB * 8 * 4);
  float*  hidden = (float*) carve(BN * cH * 4);
  __bf16* act_bf = (__bf16*)carve(BN * cH * 2);     // enc1-out / silu(u1)-out
  __bf16* ucat   = (__bf16*)carve(BN * 2 * cH * 2);
  float*  agg    = (float*) carve(BN * cH * 4);
  float*  u2out  = (float*) carve(BN * cH * 4);
  __bf16* a_tgt  = (__bf16*)carve(BE * cH * 2);
  __bf16* a_src  = (__bf16*)carve(BE * cH * 2);
  __bf16* qe     = (__bf16*)carve(BE * cH * 2);
  __bf16* ke     = (__bf16*)carve(BE * cH * 2);
  __bf16* ve     = (__bf16*)carve(BE * cH * 2);
  float*  ebl    = (float*) carve((size_t)cE * 4);
  float*  logits = (float*) carve(BE * 4);
  float*  mbuf   = (float*) carve(BN * 4);
  float*  sbuf   = (float*) carve(BN * 4);
  float*  pooled = (float*) carve(3 * cB * cH * 4);
  __bf16* wq  = (__bf16*)carve(cH * cH * 2);
  __bf16* wk  = (__bf16*)carve(cH * cH * 2);
  __bf16* wv  = (__bf16*)carve(cH * cH * 2);
  __bf16* wu1 = (__bf16*)carve(2 * cH * cH * 2);
  __bf16* wu2 = (__bf16*)carve(cH * cH * 2);
  __bf16* we2 = (__bf16*)carve(cH * cH * 2);

  auto blks = [](size_t n) { return (unsigned)((n + 255) / 256); };
  const dim3 gemm_node((unsigned)(BN / 64), cH / 64);   // 256 x 3
  const dim3 gemm_edge((unsigned)(BE / 64), cH / 64);   // 2048 x 3

  // ---- encoder
  stats_kernel<<<cB, 256, 0, stream>>>(srcm, candm, stats);
  enc1_kernel<<<blks(BN * cH), 256, 0, stream>>>(points, srcm, candm, stats,
                                                 enc1.w, enc1.b, act_bf);
  pack_w_kernel<<<blks(cH * cH), 256, 0, stream>>>(enc2.w, we2, cH, cH);
  gemm_bias_kernel<false, float><<<gemm_node, 128, 0, stream>>>(
      act_bf, we2, enc2.b, hidden, (int)BN, cH, cH);

  // ---- GAT layers
  for (int l = 0; l < cL; ++l) {
    pack_w_kernel<<<blks(cH * cH), 256, 0, stream>>>(q[l].w, wq, cH, cH);
    pack_w_kernel<<<blks(cH * cH), 256, 0, stream>>>(k[l].w, wk, cH, cH);
    pack_w_kernel<<<blks(cH * cH), 256, 0, stream>>>(v[l].w, wv, cH, cH);
    pack_w_kernel<<<blks(2 * cH * cH), 256, 0, stream>>>(u1[l].w, wu1, 2 * cH, cH);
    pack_w_kernel<<<blks(cH * cH), 256, 0, stream>>>(u2[l].w, wu2, cH, cH);

    edge_prep_kernel<<<blks(BE * cH), 256, 0, stream>>>(hidden, eidx, etype,
                                                        edge_emb[l], a_tgt, a_src);
    ebl_kernel<<<blks(cE), 256, 0, stream>>>(edge_emb[l], etype, eb[l].w, eb[l].b, ebl);

    gemm_bias_kernel<false, __bf16><<<gemm_edge, 128, 0, stream>>>(
        a_tgt, wq, q[l].b, qe, (int)BE, cH, cH);
    gemm_bias_kernel<false, __bf16><<<gemm_edge, 128, 0, stream>>>(
        a_src, wk, k[l].b, ke, (int)BE, cH, cH);
    gemm_bias_kernel<false, __bf16><<<gemm_edge, 128, 0, stream>>>(
        a_src, wv, v[l].b, ve, (int)BE, cH, cH);

    logits_kernel<<<(unsigned)(BE / 8), 256, 0, stream>>>(qe, ke, ebl, logits);

    fill_kernel<<<blks(BN), 256, 0, stream>>>(mbuf, -3.402823466e38f, (int)BN);
    fill_kernel<<<blks(BN), 256, 0, stream>>>(sbuf, 0.f, (int)BN);
    segmax_kernel<<<blks(BE), 256, 0, stream>>>(logits, eidx, mbuf);
    expsum_kernel<<<blks(BE), 256, 0, stream>>>(logits, eidx, mbuf, sbuf);

    fill_kernel<<<blks(BN * cH), 256, 0, stream>>>(agg, 0.f, (int)(BN * cH));
    agg_kernel<<<blks(BE * cH), 256, 0, stream>>>(ve, logits, sbuf, eidx, agg);

    ucat_kernel<<<blks(BN * 2 * cH), 256, 0, stream>>>(hidden, agg, ucat);
    gemm_bias_kernel<true, __bf16><<<gemm_node, 128, 0, stream>>>(
        ucat, wu1, u1[l].b, act_bf, (int)BN, 2 * cH, cH);
    gemm_bias_kernel<false, float><<<gemm_node, 128, 0, stream>>>(
        act_bf, wu2, u2[l].b, u2out, (int)BN, cH, cH);
    resln_kernel<<<(unsigned)BN, 256, 0, stream>>>(hidden, u2out, ln_g[l], ln_b[l]);
  }

  // ---- pooling + head
  pool_kernel<<<cB * cH, 256, 0, stream>>>(hidden, srcm, candm, stats, pooled);
  head_kernel<<<cB, 256, 0, stream>>>(pooled, stats, energy, head1.w, head1.b,
                                      head2.w, head2.b, (float*)d_out);
  (void)n_in; (void)out_size; (void)ws_size;
}
